// Model_17961553232512
// MI455X (gfx1250) — compile-verified, matrix operand order checked by reading, not run on hardware
//
#include <hip/hip_runtime.h>
#include <math.h>

// ---------------- model constants ----------------
// B=4, T=512, NV=64, D=512, DS=16, DFF=512, EL=2, K=3, PL=96, NH=8,
// DT_RANK=32, CONV_CH=32, SKIP_CH=32, NODE_DIM=32, GCN_DEPTH=2, alpha=0.3

typedef __attribute__((ext_vector_type(16))) __bf16 v16bf;
typedef __attribute__((ext_vector_type(8)))  float  v8f;

// pack two f32 -> packed 2x bf16 (round-to-nearest-even), element0 in low half
__device__ __forceinline__ unsigned pk2(float lo, float hi) {
  union { float f; unsigned u; } a, b; a.f = lo; b.f = hi;
  unsigned ul = a.u + (0x7FFFu + ((a.u >> 16) & 1u));
  unsigned uh = b.u + (0x7FFFu + ((b.u >> 16) & 1u));
  return (uh & 0xFFFF0000u) | (ul >> 16);
}

__device__ __forceinline__ float act_fn(float v, int act) {
  if (act == 1) return 0.5f * v * (1.f + erff(v * 0.70710678118654752f)); // exact gelu
  if (act == 2) return v / (1.f + __expf(-v));                            // silu
  if (act == 3) return (v > 20.f) ? v : log1pf(__expf(v));                // softplus
  return v;
}

// f32 buffer -> packed bf16 shadow buffer (one conversion pass, reused by GEMMs)
__global__ void k_f2bf(const float* src, unsigned short* dst, long long n) {
  long long i = ((long long)blockIdx.x * 256 + threadIdx.x) * 2;
  if (i >= n) return;
  if (i + 1 < n) {
    *(unsigned*)(dst + i) = pk2(src[i], src[i + 1]);
  } else {
    union { float f; unsigned u; } a; a.f = src[i];
    unsigned u = a.u + (0x7FFFu + ((a.u >> 16) & 1u));
    dst[i] = (unsigned short)(u >> 16);
  }
}

// load one 16x32 bf16 fragment row-slice for this lane.
// base = row start; halfs 0..7 = k koff..koff+7, halfs 8..15 = koff+16..koff+23.
template <bool CONTIG>
__device__ __forceinline__ v16bf load_frag(const unsigned short* base, int koff, long long sk) {
  union { v16bf v; unsigned u[8]; uint4 q[2]; } r;
  if (CONTIG) {
    r.q[0] = *(const uint4*)(base + koff);
    r.q[1] = *(const uint4*)(base + koff + 16);
  } else {
#pragma unroll
    for (int j = 0; j < 4; ++j) {
      unsigned l0 = base[(long long)(koff + 2 * j) * sk];
      unsigned h0 = base[(long long)(koff + 2 * j + 1) * sk];
      unsigned l1 = base[(long long)(koff + 16 + 2 * j) * sk];
      unsigned h1 = base[(long long)(koff + 17 + 2 * j) * sk];
      r.u[j]     = l0 | (h0 << 16);
      r.u[4 + j] = l1 | (h1 << 16);
    }
  }
  return r.v;
}

// ---------------- generalized strided-batched WMMA GEMM (bf16 in, f32 out) ----
// C[b, m, n] = act( scale * sum_k A[b,m,k] * W[b,n,k] + bias[n] )
// batch index z -> (b0 = z/bdiv, b1 = z%bdiv). K must be a multiple of 32.
// One wave computes a 16x(16*NT) output strip, reusing the A fragment NT times.
struct GemmArgs {
  const unsigned short* A; const unsigned short* W; const float* bias; float* C;
  int M, N, K, act, bdiv; float scale;
  long long sAb0, sAb1, sAm, sAk;
  long long sWb0, sWb1, sWn, sWk;
  long long sCb0, sCb1, sCm, sCn;
};

template <int NT, bool AC, bool WC>
__global__ void k_gemm_t(GemmArgs g) {
  int wave = threadIdx.x >> 5, lane = threadIdx.x & 31;
  int tile_m = blockIdx.y * 8 + wave;
  if (tile_m * 16 >= g.M) return;          // wave-uniform: EXEC stays all-ones
  int tile_n0 = blockIdx.x * NT;
  int z  = blockIdx.z;
  int b0 = z / g.bdiv, b1 = z % g.bdiv;
  int mrow = lane & 15;
  int kb   = (lane >> 4) << 3;             // 0 or 8 : K-half per half-wave
  const unsigned short* Ab  = g.A + g.sAb0 * b0 + g.sAb1 * b1 +
                              (long long)(tile_m * 16 + mrow) * g.sAm;
  const unsigned short* Wb0 = g.W + g.sWb0 * b0 + g.sWb1 * b1 +
                              (long long)(tile_n0 * 16 + mrow) * g.sWn;
  v8f acc[NT] = {};
  for (int k0 = 0; k0 < g.K; k0 += 32) {
    v16bf av = load_frag<AC>(Ab, k0 + kb, g.sAk);
#pragma unroll
    for (int t = 0; t < NT; ++t) {
      v16bf bv = load_frag<WC>(Wb0 + (long long)t * 16 * g.sWn, k0 + kb, g.sWk);
      acc[t] = __builtin_amdgcn_wmma_f32_16x16x32_bf16(false, av, false, bv,
                                                       (short)0, acc[t], false, false);
    }
  }
  float* Cb = g.C + g.sCb0 * b0 + g.sCb1 * b1;
  int rhi = (lane >> 4) << 3;
#pragma unroll
  for (int t = 0; t < NT; ++t) {
    int col = (tile_n0 + t) * 16 + mrow;
    float bb = g.bias ? g.bias[col] : 0.f;
#pragma unroll
    for (int r = 0; r < 8; ++r) {
      int row = tile_m * 16 + rhi + r;
      float v = act_fn(g.scale * acc[t][r] + bb, g.act);
      Cb[(long long)row * g.sCm + (long long)col * g.sCn] = v;
    }
  }
}

// ---------------- elementwise / reduction kernels ----------------
__global__ void k_copy(float* dst, const float* src, long long n) {
  long long i = (long long)blockIdx.x * 256 + threadIdx.x;
  if (i < n) dst[i] = src[i];
}
__global__ void k_add2(const float* a, const float* b, float* c, long long n) {
  long long i = (long long)blockIdx.x * 256 + threadIdx.x;
  if (i < n) c[i] = a[i] + b[i];
}
__global__ void k_act(float* x, long long n, int act) {
  long long i = (long long)blockIdx.x * 256 + threadIdx.x;
  if (i < n) x[i] = act_fn(x[i], act);
}

// per-(b,n) mean/std over T + normalized xn   (x: [B,T,NV])
__global__ void k_stats(const float* x, float* xn, float* means, float* stdv) {
  int bn = blockIdx.x; int b = bn >> 6, n = bn & 63; int t = threadIdx.x;
  const float* base = x + (long long)b * 512 * 64 + n;
  float v0 = base[(long long)t * 64];
  float v1 = base[(long long)(t + 256) * 64];
  __shared__ float sh[256];
  sh[t] = v0 + v1; __syncthreads();
  for (int s = 128; s > 0; s >>= 1) { if (t < s) sh[t] += sh[t + s]; __syncthreads(); }
  float mean = sh[0] / 512.f; __syncthreads();
  float d0 = v0 - mean, d1 = v1 - mean;
  sh[t] = d0 * d0 + d1 * d1; __syncthreads();
  for (int s = 128; s > 0; s >>= 1) { if (t < s) sh[t] += sh[t + s]; __syncthreads(); }
  float sd = sqrtf(sh[0] / 512.f + 1e-5f);
  float is = 1.f / sd;
  float* xb = xn + (long long)b * 512 * 64 + n;
  xb[(long long)t * 64]         = d0 * is;
  xb[(long long)(t + 256) * 64] = d1 * is;
  if (t == 0) { means[bn] = mean; stdv[bn] = sd; }
}

// tok_w (D,NV,3) -> tokw2 (D, 3*NV) laid out [tap*64+n]
__global__ void k_repack_tokw(const float* tw, float* tw2) {
  int i = blockIdx.x * 256 + threadIdx.x;
  if (i >= 512 * 192) return;
  int d = i / 192, j = i % 192, tap = j / 64, n = j % 64;
  tw2[i] = tw[d * 192 + n * 3 + tap];
}
// xg[b,t, tap*64+n] = xn[b, (t+tap-1) mod T, n]   (circular pad k=3)
__global__ void k_build_xg(const float* xn, float* xg) {
  long long i = (long long)blockIdx.x * 256 + threadIdx.x;
  if (i >= (long long)4 * 512 * 192) return;
  int j = (int)(i % 192); long long rem = i / 192;
  int t = (int)(rem % 512); int b = (int)(rem / 512);
  int tap = j / 64, n = j % 64;
  int ts = (t + tap - 1 + 512) & 511;
  xg[i] = xn[((long long)b * 512 + ts) * 64 + n];
}
__global__ void k_addpos(float* ex, long long n) {
  long long i = (long long)blockIdx.x * 256 + threadIdx.x;
  if (i >= n) return;
  int d = (int)(i & 511); int t = (int)((i >> 9) & 511);
  float freq = __expf(-(float)(d & ~1) * (9.210340371976184f / 512.f));
  float ang = (float)t * freq;
  ex[i] += (d & 1) ? __cosf(ang) : __sinf(ang);
}

// amp mean over D of |rfft(enc_x)| : one block per (b,f)
__global__ void k_dft(const float* ex, float* ampmean) {
  int bf = blockIdx.x; int b = bf / 257, f = bf % 257; int t = threadIdx.x;
  float w = -6.283185307179586f * (float)f / 512.f;
  float loc = 0.f;
  for (int d = t; d < 512; d += 256) {
    float re = 0.f, im = 0.f;
    const float* xb = ex + (long long)b * 512 * 512 + d;
    for (int tt = 0; tt < 512; ++tt) {
      float sv, cv; __sincosf(w * (float)tt, &sv, &cv);
      float v = xb[(long long)tt * 512];
      re += v * cv; im += v * sv;
    }
    loc += sqrtf(re * re + im * im);
  }
  __shared__ float sh[256];
  sh[t] = loc; __syncthreads();
  for (int s = 128; s > 0; s >>= 1) { if (t < s) sh[t] += sh[t + s]; __syncthreads(); }
  if (t == 0) ampmean[bf] = sh[0] / 512.f;
}

// single-thread: top-3 freqs -> scale info + squared softmax weights
__global__ void k_topk(const float* am, int* sinfo, float* swsq) {
  if (threadIdx.x != 0 || blockIdx.x != 0) return;
  float fl[257]; int idx[3];
  for (int f = 0; f < 257; ++f)
    fl[f] = 0.25f * (am[f] + am[257 + f] + am[514 + f] + am[771 + f]);
  fl[0] = 0.f;
  for (int k = 0; k < 3; ++k) {
    int best = 1; float bv = -1e30f;
    for (int f = 0; f < 257; ++f) if (fl[f] > bv) { bv = fl[f]; best = f; }
    idx[k] = best; fl[best] = -1e31f;
  }
  for (int k = 0; k < 3; ++k) {
    int f = idx[k] < 1 ? 1 : idx[k];
    int s = 512 / f; int rows = (512 + s - 1) / s; int len = rows * s;
    sinfo[k * 4] = f; sinfo[k * 4 + 1] = s; sinfo[k * 4 + 2] = rows; sinfo[k * 4 + 3] = len;
  }
  for (int b = 0; b < 4; ++b) {
    float w[3], mx = -1e30f;
    for (int k = 0; k < 3; ++k) { w[k] = am[b * 257 + idx[k]]; mx = fmaxf(mx, w[k]); }
    float ss = 0.f;
    for (int k = 0; k < 3; ++k) { w[k] = __expf(w[k] - mx); ss += w[k]; }
    for (int k = 0; k < 3; ++k) { float p = w[k] / ss; swsq[b * 4 + k] = p * p; }
  }
}

// enc_x [B,T,D] -> o4 [B,D,len] (image rows x s, zero padded past T)
__global__ void k_sprep(const float* ex, float* o4, const int* sinfo, int kk, long long n) {
  long long i = (long long)blockIdx.x * 256 + threadIdx.x;
  if (i >= n) return;
  int p = (int)(i & 1023); long long rem = i >> 10;
  int d = (int)(rem & 511); int b = (int)(rem >> 9);
  int len = sinfo[kk * 4 + 3];
  if (p >= len) return;
  o4[i] = (p < 512) ? ex[((long long)(b * 512 + p)) * 512 + d] : 0.f;
}

// inception: (conv1x1+conv3x3+conv5x5)/3 with SAME padding over [rows,s]
__global__ void k_incep(const float* in, float* out,
                        const float* w1, const float* b1, const float* w3, const float* b3,
                        const float* w5, const float* b5,
                        int Cin, int Cout, const int* sinfo, int kk) {
  long long i = (long long)blockIdx.x * 256 + threadIdx.x;
  long long total = (long long)4 * Cout * 1024;
  if (i >= total) return;
  int p = (int)(i & 1023); long long rem = i >> 10;
  int co = (int)(rem % Cout); int b = (int)(rem / Cout);
  int s = sinfo[kk * 4 + 1], rows = sinfo[kk * 4 + 2], len = sinfo[kk * 4 + 3];
  if (p >= len) return;
  int r = p / s, c = p % s;
  float acc = b1[co] + b3[co] + b5[co];
  const float* inb = in + (long long)b * Cin * 1024;
  {
    const float* wv = w1 + (long long)co * Cin;
    for (int ci = 0; ci < Cin; ++ci) acc += inb[(long long)ci * 1024 + p] * wv[ci];
  }
  {
    const float* wv = w3 + (long long)co * Cin * 9;
    for (int ci = 0; ci < Cin; ++ci) {
      const float* ib = inb + (long long)ci * 1024;
      for (int kh = 0; kh < 3; ++kh) {
        int rr = r + kh - 1; if (rr < 0 || rr >= rows) continue;
        for (int kw = 0; kw < 3; ++kw) {
          int cc = c + kw - 1; if (cc < 0 || cc >= s) continue;
          acc += ib[rr * s + cc] * wv[(long long)ci * 9 + kh * 3 + kw];
        }
      }
    }
  }
  {
    const float* wv = w5 + (long long)co * Cin * 25;
    for (int ci = 0; ci < Cin; ++ci) {
      const float* ib = inb + (long long)ci * 1024;
      for (int kh = 0; kh < 5; ++kh) {
        int rr = r + kh - 2; if (rr < 0 || rr >= rows) continue;
        for (int kw = 0; kw < 5; ++kw) {
          int cc = c + kw - 2; if (cc < 0 || cc >= s) continue;
          acc += ib[rr * s + cc] * wv[(long long)ci * 25 + kh * 5 + kw];
        }
      }
    }
  }
  out[i] = acc * (1.f / 3.f);
}

__global__ void k_unpack(const float* o4, float* br, long long n) {
  long long i = (long long)blockIdx.x * 256 + threadIdx.x;
  if (i >= n) return;
  int d = (int)(i & 511); long long rem = i >> 9;
  int t = (int)(rem & 511); int b = (int)(rem >> 9);
  br[i] = o4[((long long)(b * 512 + d)) * 1024 + t];
}
__global__ void k_accum(float* outm, const float* br, const float* swsq, int kk, long long n) {
  long long i = (long long)blockIdx.x * 256 + threadIdx.x;
  if (i >= n) return;
  int b = (int)(i >> 18);
  outm[i] += br[i] * swsq[b * 4 + kk];
}

// mamba depthwise causal conv (k=2) + bias + silu; rev flips direction
__global__ void k_dwconv(const float* xz, const float* cw, const float* cb, float* xc,
                         int L, int rev, long long n) {
  long long i = (long long)blockIdx.x * 256 + threadIdx.x;
  if (i >= n) return;
  int d = (int)(i & 511); long long m = i >> 9;
  int l = (int)(m % (long long)L);
  float cur = xz[m * 1024 + d];
  float prev = 0.f;
  if (rev) { if (l < L - 1) prev = xz[(m + 1) * 1024 + d]; }
  else     { if (l > 0)     prev = xz[(m - 1) * 1024 + d]; }
  float v = cw[d * 2] * prev + cw[d * 2 + 1] * cur + cb[d];
  xc[i] = v / (1.f + __expf(-v));
}

// selective scan: per (b,d) 16-state recurrence; output fused with silu(z)
__global__ void k_scan(const float* xc, const float* xdbl, const float* dtf,
                       const float* A_log, const float* Dp, const float* xz,
                       float* ym, int L, int rev) {
  int idx = blockIdx.x * 256 + threadIdx.x;
  if (idx >= 4 * 512) return;
  int d = idx & 511, b = idx >> 9;
  float A[16], h[16];
#pragma unroll
  for (int s = 0; s < 16; ++s) { A[s] = -__expf(A_log[d * 16 + s]); h[s] = 0.f; }
  for (int li = 0; li < L; ++li) {
    int l = rev ? (L - 1 - li) : li;
    long long m = (long long)b * L + l;
    float dt = dtf[m * 512 + d];
    float x  = xc[m * 512 + d];
    const float* bd = xdbl + m * 64;
    float y = 0.f;
#pragma unroll
    for (int s = 0; s < 16; ++s) {
      float dA = __expf(dt * A[s]);
      h[s] = dA * h[s] + dt * bd[32 + s] * x;
      y += h[s] * bd[48 + s];
    }
    y += x * Dp[d];
    float z = xz[m * 1024 + 512 + d];
    ym[m * 512 + d] = y * (z / (1.f + __expf(-z)));
  }
}

// layernorm over D=512, optional fused residual (out = LN(x [+ r]) * g + b)
__global__ void k_ln(const float* x, const float* r, const float* g, const float* bta, float* out) {
  long long row = blockIdx.x; int t = threadIdx.x;
  const float* xr = x + row * 512;
  float v0 = xr[t], v1 = xr[t + 256];
  if (r) { const float* rr = r + row * 512; v0 += rr[t]; v1 += rr[t + 256]; }
  __shared__ float sh[256];
  sh[t] = v0 + v1; __syncthreads();
  for (int s = 128; s > 0; s >>= 1) { if (t < s) sh[t] += sh[t + s]; __syncthreads(); }
  float mean = sh[0] / 512.f; __syncthreads();
  float d0 = v0 - mean, d1 = v1 - mean;
  sh[t] = d0 * d0 + d1 * d1; __syncthreads();
  for (int s = 128; s > 0; s >>= 1) { if (t < s) sh[t] += sh[t + s]; __syncthreads(); }
  float rs = rsqrtf(sh[0] / 512.f + 1e-5f);
  out[row * 512 + t]       = d0 * rs * g[t]       + bta[t];
  out[row * 512 + t + 256] = d1 * rs * g[t + 256] + bta[t + 256];
}

__global__ void k_softmax(float* x, int cols) {
  long long row = blockIdx.x; int t = threadIdx.x;
  float* xr = x + row * (long long)cols;
  __shared__ float sh[256];
  float m = -1e30f;
  for (int c = t; c < cols; c += 256) m = fmaxf(m, xr[c]);
  sh[t] = m; __syncthreads();
  for (int s = 128; s > 0; s >>= 1) { if (t < s) sh[t] = fmaxf(sh[t], sh[t + s]); __syncthreads(); }
  float mx = sh[0]; __syncthreads();
  float su = 0.f;
  for (int c = t; c < cols; c += 256) { float e = __expf(xr[c] - mx); xr[c] = e; su += e; }
  sh[t] = su; __syncthreads();
  for (int s = 128; s > 0; s >>= 1) { if (t < s) sh[t] += sh[t + s]; __syncthreads(); }
  float inv = 1.f / sh[0];
  for (int c = t; c < cols; c += 256) xr[c] *= inv;
}

// graph adjacency: a = rownorm(softmax_row(relu(n1@n2)) + I)
__global__ void k_gadj(const float* n1, const float* n2, float* a) {
  int i = threadIdx.x; if (i >= 64) return;
  float row[64]; float mx = -1e30f;
  for (int j = 0; j < 64; ++j) {
    float dsum = 0.f;
    for (int k = 0; k < 32; ++k) dsum += n1[i * 32 + k] * n2[k * 64 + j];
    float v = fmaxf(dsum, 0.f); row[j] = v; mx = fmaxf(mx, v);
  }
  float ss = 0.f;
  for (int j = 0; j < 64; ++j) { row[j] = __expf(row[j] - mx); ss += row[j]; }
  for (int j = 0; j < 64; ++j) row[j] /= ss;
  row[i] += 1.f;
  float rs = 0.f;
  for (int j = 0; j < 64; ++j) rs += row[j];
  for (int j = 0; j < 64; ++j) a[i * 64 + j] = row[j] / rs;
}

// start conv: gho[b,c,h,w] = sum_i enc_out[b,w,h+i]*sw[c,i] + sb[c]  (449 taps)
__global__ void k_startconv(const float* eo, const float* sw, const float* sb, float* gho) {
  long long i = (long long)blockIdx.x * 256 + threadIdx.x;
  if (i >= (long long)4 * 32 * 64 * 64) return;
  int w = (int)(i & 63); long long rem = i >> 6;
  int h = (int)(rem & 63); rem >>= 6;
  int c = (int)(rem & 31); int b = (int)(rem >> 5);
  float acc = sb[c];
  const float* xb = eo + ((long long)b * 64 + w) * 512 + h;
  const float* wv = sw + (long long)c * 449;
  for (int ii = 0; ii < 449; ++ii) acc += xb[ii] * wv[ii];
  gho[(long long)b * 393216 + (long long)c * 4096 + h * 64 + w] = acc;
}

// GCN propagate: slice sout = 0.3*slice0 + 0.7 * einsum(slice sin, a)
__global__ void k_prop(float* gho, const float* a, int sin_, int sout) {
  long long i = (long long)blockIdx.x * 256 + threadIdx.x;
  if (i >= (long long)4 * 32 * 4096) return;
  int l = (int)(i & 63); long long rem = i >> 6;
  int v = (int)(rem & 63); rem >>= 6;
  int c = (int)(rem & 31); int b = (int)(rem >> 5);
  const float* gb = gho + (long long)b * 393216;
  const float* hin = gb + (long long)sin_ * 131072 + (long long)c * 4096 + l;
  const float* ar = a + v * 64;
  float acc = 0.f;
  for (int w = 0; w < 64; ++w) acc += hin[(long long)w * 64] * ar[w];
  float o = gb[(long long)c * 4096 + v * 64 + l];
  gho[(long long)b * 393216 + (long long)(sout * 32 + c) * 4096 + v * 64 + l] =
      0.3f * o + 0.7f * acc;
}

// end conv: ge[b,o,v] = sum_{c,l} ghm[b,c,v,l] * ew[o,c,l] + eb[o]
__global__ void k_endconv(const float* ghm, const float* ew, const float* eb, float* ge) {
  long long i = (long long)blockIdx.x * 256 + threadIdx.x;
  if (i >= (long long)4 * 64 * 64) return;
  int v = (int)(i & 63); long long rem = i >> 6;
  int o = (int)(rem & 63); int b = (int)(rem >> 6);
  float acc = eb[o];
  const float* hb = ghm + (long long)b * 131072 + (long long)v * 64;
  const float* wb = ew + (long long)o * 2048;
  for (int c = 0; c < 32; ++c)
    for (int l = 0; l < 64; ++l)
      acc += hb[(long long)c * 4096 + l] * wb[c * 64 + l];
  ge[(long long)b * 4096 + o * 64 + v] = acc;
}

// dec_cat[b,v,:512]=a_out[b,v,:] ; [512:]=b_lin[b,t,v] transposed
__global__ void k_deccat(const float* aoutb, const float* blin, float* dc, long long n) {
  long long i = (long long)blockIdx.x * 256 + threadIdx.x;
  if (i >= n) return;
  int j = (int)(i & 1023); long long rem = i >> 10;
  int v = (int)(rem & 63); int b = (int)(rem >> 6);
  dc[i] = (j < 512) ? aoutb[((long long)b * 64 + v) * 512 + j]
                    : blin[((long long)b * 512 + (j - 512)) * 64 + v];
}

__global__ void k_final(const float* dec2, const float* stdv, const float* means,
                        float* out, long long n) {
  long long i = (long long)blockIdx.x * 256 + threadIdx.x;
  if (i >= n) return;
  int nn = (int)(i & 63); long long rem = i >> 6;
  int pl = (int)(rem % 96); int b = (int)(rem / 96);
  out[i] = dec2[((long long)b * 64 + nn) * 96 + pl] * stdv[b * 64 + nn] + means[b * 64 + nn];
}

// ---------------- host helpers ----------------
template <int NT>
static void gemm_disp(hipStream_t st, const GemmArgs& g, dim3 grid, bool ac, bool wc) {
  if (ac && wc)       k_gemm_t<NT, true,  true ><<<grid, 256, 0, st>>>(g);
  else if (ac)        k_gemm_t<NT, true,  false><<<grid, 256, 0, st>>>(g);
  else if (wc)        k_gemm_t<NT, false, true ><<<grid, 256, 0, st>>>(g);
  else                k_gemm_t<NT, false, false><<<grid, 256, 0, st>>>(g);
}

static void gemm_full(hipStream_t st, const unsigned short* A, const unsigned short* W,
                      const float* bias, float* C,
                      int M, int N, int K, int batch, int bdiv, float scale, int act,
                      long long sAb0, long long sAb1, long long sAm, long long sAk,
                      long long sWb0, long long sWb1, long long sWn, long long sWk,
                      long long sCb0, long long sCb1, long long sCm, long long sCn) {
  GemmArgs g{A, W, bias, C, M, N, K, act, bdiv, scale,
             sAb0, sAb1, sAm, sAk, sWb0, sWb1, sWn, sWk, sCb0, sCb1, sCm, sCn};
  bool ac = (sAk == 1), wc = (sWk == 1);
  int nt = (N % 64 == 0) ? 4 : ((N % 32 == 0) ? 2 : 1);
  dim3 grid((unsigned)(N / (16 * nt)), (unsigned)((M / 16 + 7) / 8), (unsigned)batch);
  if (nt == 4)      gemm_disp<4>(st, g, grid, ac, wc);
  else if (nt == 2) gemm_disp<2>(st, g, grid, ac, wc);
  else              gemm_disp<1>(st, g, grid, ac, wc);
}
static void gemm_rm(hipStream_t st, const unsigned short* A, const unsigned short* W,
                    const float* bias, float* C, int M, int N, int K, int act) {
  gemm_full(st, A, W, bias, C, M, N, K, 1, 1, 1.f, act,
            0, 0, K, 1, 0, 0, K, 1, 0, 0, N, 1);
}
static inline unsigned GB(long long n) { return (unsigned)((n + 255) / 256); }

// parameter index map (flattened setup_inputs order)
// 0 x_enc, 1 tok_w, 2 inv_w, 3 inv_b, 4..57 enc0, 58..111 enc1,
// 112-114 incep1_w, 115-117 incep1_b, 118-120 incep2_w, 121-123 incep2_b,
// 124..135 graph, 136..143 ca1, 144..151 ca2, 152 c_lin_w, 153 c_lin_b,
// 154 cat_w, 155 cat_b, 156 proj_w, 157 proj_b

extern "C" void kernel_launch(void* const* d_in, const int* in_sizes, int n_in,
                              void* d_out, int out_size, void* d_ws, size_t ws_size,
                              hipStream_t stream) {
  (void)out_size; (void)ws_size;
  auto P = [&](int i) { return (const float*)d_in[i]; };
  char* wp = (char*)d_ws;
  auto AF = [&](size_t nf) {
    float* r = (float*)wp;
    wp += ((nf * sizeof(float) + 255) / 256) * 256;
    return r;
  };
  auto AB = [&](size_t nf) {
    unsigned short* r = (unsigned short*)wp;
    wp += ((nf * sizeof(unsigned short) + 255) / 256) * 256;
    return r;
  };

  float* xn      = AF(131072);
  float* means   = AF(256);
  float* stdv    = AF(256);
  float* tokw2   = AF(98304);
  float* xg      = AF(393216);
  float* enc_x   = AF(1048576);
  float* ampmean = AF(1028);
  float* swsq    = AF(16);
  int*   sinfo   = (int*)AF(16);
  float* out_c_m = AF(1048576);
  float* o4a     = AF(2097152);
  float* o4b     = AF(131072);
  float* branch  = AF(1048576);
  float* mxz     = AF(2097152);
  float* mxc     = AF(1048576);
  float* mxd     = AF(131072);
  float* mdt     = AF(1048576);
  float* mym     = AF(1048576);
  float* em1     = AF(1048576);
  float* em2     = AF(1048576);
  float* exl     = AF(1048576);
  float* enc_out = AF(131072);
  float* gho     = AF(1572864);
  float* ghm     = AF(524288);
  float* ge      = AF(16384);
  float* adjA    = AF(4096);
  float* og      = AF(131072);
  float* glin    = AF(131072);
  float* aq      = AF(1048576);
  float* ak      = AF(1048576);
  float* avv     = AF(1048576);
  float* asc     = AF(1048576);
  float* ao      = AF(1048576);
  float* ap      = AF(1048576);
  float* aoutb   = AF(131072);
  float* b_out   = AF(1048576);
  float* blin    = AF(131072);
  float* dcat    = AF(262144);
  float* dec1    = AF(131072);
  float* dec2    = AF(24576);

  // bf16 shadow buffers (GEMM operands)
  unsigned short* bxg    = AB(393216);
  unsigned short* btokw2 = AB(98304);
  unsigned short* bxn    = AB(131072);
  unsigned short* bx     = AB(1048576);
  unsigned short* bmxc   = AB(1048576);
  unsigned short* bmxd   = AB(131072);
  unsigned short* bmym   = AB(1048576);
  unsigned short* bexl   = AB(1048576);
  unsigned short* bmdt   = AB(1048576);
  unsigned short* bgho   = AB(1572864);
  unsigned short* bge    = AB(16384);
  unsigned short* baq    = AB(1048576);
  unsigned short* bak    = AB(1048576);
  unsigned short* bavv   = AB(1048576);
  unsigned short* basc   = AB(1048576);
  unsigned short* bao    = AB(1048576);
  unsigned short* bocm   = AB(1048576);
  unsigned short* bog    = AB(131072);
  unsigned short* bbout  = AB(1048576);
  unsigned short* bdcat  = AB(262144);
  unsigned short* bdec1  = AB(131072);

  // activation f32 -> bf16 one-shot conversion
  auto CA = [&](const float* src, unsigned short* dst, long long n) {
    k_f2bf<<<GB((n + 1) / 2), 256, 0, stream>>>(src, dst, n);
    return (const unsigned short*)dst;
  };
  // weight conversion, memoized per parameter index (deterministic order)
  unsigned short* wcache[158] = {};
  auto CW = [&](int i) -> const unsigned short* {
    if (!wcache[i]) {
      long long n = (long long)in_sizes[i];
      wcache[i] = AB((size_t)n);
      k_f2bf<<<GB((n + 1) / 2), 256, 0, stream>>>(P(i), wcache[i], n);
    }
    return wcache[i];
  };

  // ---- mamba block (input already bf16) ----
  auto run_mamba = [&](const unsigned short* xb, int mb, int L, int rev, float* out) {
    int M = 4 * L;
    gemm_rm(stream, xb, CW(mb + 0), nullptr, mxz, M, 1024, 512, 0);          // in_proj
    long long n1 = (long long)M * 512;
    k_dwconv<<<GB(n1), 256, 0, stream>>>(mxz, P(mb + 1), P(mb + 2), mxc, L, rev, n1);
    CA(mxc, bmxc, n1);
    gemm_rm(stream, bmxc, CW(mb + 3), nullptr, mxd, M, 64, 512, 0);          // x_proj
    CA(mxd, bmxd, (long long)M * 64);
    gemm_full(stream, bmxd, CW(mb + 4), P(mb + 5), mdt, M, 512, 32, 1, 1, 1.f, 3,
              0, 0, 64, 1, 0, 0, 32, 1, 0, 0, 512, 1);                       // dt_proj+softplus
    k_scan<<<GB(4 * 512), 256, 0, stream>>>(mxc, mxd, mdt, P(mb + 6), P(mb + 7),
                                            mxz, mym, L, rev);
    CA(mym, bmym, n1);
    gemm_rm(stream, bmym, CW(mb + 8), nullptr, out, M, 512, 512, 0);         // out_proj
  };

  // ---- encoder (2 layers + final LN), x modified in place ----
  auto run_encoder = [&](float* x, int eb, int L) {
    int M = 4 * L;
    long long n = (long long)M * 512;
    for (int l = 0; l < 2; ++l) {
      int lb = eb + l * 26;
      CA(x, bx, n);
      run_mamba(bx, lb + 0, L, 0, em1);
      run_mamba(bx, lb + 9, L, 1, em2);
      k_add2<<<GB(n), 256, 0, stream>>>(em1, em2, em1, n);
      k_ln<<<M, 256, 0, stream>>>(x, em1, P(lb + 22), P(lb + 23), exl);
      CA(exl, bexl, n);
      gemm_rm(stream, bexl, CW(lb + 18), P(lb + 19), mdt, M, 512, 512, 1);   // FFN1 + gelu
      CA(mdt, bmdt, n);
      gemm_rm(stream, bmdt, CW(lb + 20), P(lb + 21), em2, M, 512, 512, 0);   // FFN2
      k_ln<<<M, 256, 0, stream>>>(exl, em2, P(lb + 24), P(lb + 25), x);
    }
    k_ln<<<M, 256, 0, stream>>>(x, nullptr, P(eb + 52), P(eb + 53), x);
  };

  // ---- cross attention -> outp = proj(attn(q_in, kv)) ----
  auto run_attn = [&](const unsigned short* qb, int Lq, const unsigned short* kvb, int Lk,
                      int ab, float* outp) {
    int Mq = 4 * Lq, Mk = 4 * Lk;
    gemm_rm(stream, qb,  CW(ab + 0), P(ab + 1), aq,  Mq, 512, 512, 0);
    gemm_rm(stream, kvb, CW(ab + 2), P(ab + 3), ak,  Mk, 512, 512, 0);
    gemm_rm(stream, kvb, CW(ab + 4), P(ab + 5), avv, Mk, 512, 512, 0);
    CA(aq, baq, (long long)Mq * 512);
    CA(ak, bak, (long long)Mk * 512);
    CA(avv, bavv, (long long)Mk * 512);
    gemm_full(stream, baq, bak, nullptr, asc, Lq, Lk, 64, 4 * 8, 8, 0.125f, 0,
              (long long)Lq * 512, 64, 512, 1,
              (long long)Lk * 512, 64, 512, 1,
              (long long)8 * Lq * Lk, (long long)Lq * Lk, Lk, 1);
    k_softmax<<<4 * 8 * Lq, 256, 0, stream>>>(asc, Lk);
    CA(asc, basc, (long long)4 * 8 * Lq * Lk);
    gemm_full(stream, basc, bavv, nullptr, ao, Lq, 64, Lk, 4 * 8, 8, 1.f, 0,
              (long long)8 * Lq * Lk, (long long)Lq * Lk, Lk, 1,
              (long long)Lk * 512, 64, 1, 512,
              (long long)Lq * 512, 64, 512, 1);
    CA(ao, bao, (long long)Mq * 512);
    gemm_rm(stream, bao, CW(ab + 6), P(ab + 7), outp, Mq, 512, 512, 0);
  };

  // ================= forward =================
  const float* X = P(0);
  long long nBTD = (long long)4 * 512 * 512;

  // embed: stats + token conv (as GEMM) + positional embedding
  k_stats<<<256, 256, 0, stream>>>(X, xn, means, stdv);
  k_repack_tokw<<<GB(512 * 192), 256, 0, stream>>>(P(1), tokw2);
  k_build_xg<<<GB((long long)4 * 512 * 192), 256, 0, stream>>>(xn, xg);
  CA(xg, bxg, (long long)4 * 512 * 192);
  CA(tokw2, btokw2, 512 * 192);
  gemm_rm(stream, bxg, btokw2, nullptr, enc_x, 2048, 512, 192, 0);
  k_addpos<<<GB(nBTD), 256, 0, stream>>>(enc_x, nBTD);

  // periods (DFT amplitudes, top-k, softmax^2 weights)
  k_dft<<<4 * 257, 256, 0, stream>>>(enc_x, ampmean);
  k_topk<<<1, 1, 0, stream>>>(ampmean, sinfo, swsq);

  // multi-scale branches
  k_copy<<<GB(nBTD), 256, 0, stream>>>(out_c_m, enc_x, nBTD);
  for (int kk = 0; kk < 3; ++kk) {
    long long n = (long long)4 * 512 * 1024;
    k_sprep<<<GB(n), 256, 0, stream>>>(enc_x, o4a, sinfo, kk, n);
    long long n2 = (long long)4 * 32 * 1024;
    k_incep<<<GB(n2), 256, 0, stream>>>(o4a, o4b, P(112), P(115), P(113), P(116),
                                        P(114), P(117), 512, 32, sinfo, kk);
    k_act<<<GB(n2), 256, 0, stream>>>(o4b, n2, 1);
    k_incep<<<GB(n), 256, 0, stream>>>(o4b, o4a, P(118), P(121), P(119), P(122),
                                       P(120), P(123), 32, 512, sinfo, kk);
    k_unpack<<<GB(nBTD), 256, 0, stream>>>(o4a, branch, nBTD);
    run_encoder(branch, 4, 512);
    k_accum<<<GB(nBTD), 256, 0, stream>>>(out_c_m, branch, swsq, kk, nBTD);
  }

  // inverted embedding: enc_out[b,n,d] = xn[b,:,n] @ inv_w[d,:] + inv_b
  CA(xn, bxn, 131072);
  gemm_full(stream, bxn, CW(2), P(3), enc_out, 64, 512, 512, 4, 1, 1.f, 0,
            32768, 0, 1, 64, 0, 0, 512, 1, 32768, 0, 512, 1);

  // graph block
  k_gadj<<<1, 64, 0, stream>>>(P(124), P(125), adjA);
  k_startconv<<<GB((long long)4 * 32 * 64 * 64), 256, 0, stream>>>(enc_out, P(126), P(127), gho);
  k_prop<<<GB((long long)4 * 32 * 4096), 256, 0, stream>>>(gho, adjA, 0, 1);
  k_prop<<<GB((long long)4 * 32 * 4096), 256, 0, stream>>>(gho, adjA, 1, 2);
  CA(gho, bgho, 1572864);
  gemm_full(stream, bgho, CW(128), P(129), ghm, 4096, 32, 96, 4, 1, 1.f, 1,
            393216, 0, 1, 4096, 0, 0, 96, 1, 131072, 0, 1, 4096);           // mlp + gelu
  k_endconv<<<GB((long long)4 * 64 * 64), 256, 0, stream>>>(ghm, P(130), P(131), ge);
  CA(ge, bge, 16384);
  gemm_full(stream, bge, CW(132), P(133), glin, 64, 512, 64, 4, 1, 1.f, 0,
            4096, 0, 64, 1, 0, 0, 64, 1, 32768, 0, 512, 1);                 // lin
  k_ln<<<256, 256, 0, stream>>>(enc_out, glin, P(134), P(135), og);

  // encoder over graph output, residual with enc_out
  run_encoder(og, 58, 64);
  k_add2<<<GB((long long)4 * 64 * 512), 256, 0, stream>>>(og, enc_out, og,
                                                          (long long)4 * 64 * 512);

  // cross attentions
  CA(out_c_m, bocm, nBTD);
  CA(og, bog, (long long)4 * 64 * 512);
  run_attn(bog, 64, bocm, 512, 136, ap);
  k_add2<<<GB((long long)4 * 64 * 512), 256, 0, stream>>>(ap, og, aoutb,
                                                          (long long)4 * 64 * 512);
  run_attn(bocm, 512, bog, 64, 144, ap);
  k_add2<<<GB(nBTD), 256, 0, stream>>>(ap, out_c_m, b_out, nBTD);

  // head
  CA(b_out, bbout, nBTD);
  gemm_rm(stream, bbout, CW(152), P(153), blin, 2048, 64, 512, 0);          // c_lin
  long long nc = (long long)4 * 64 * 1024;
  k_deccat<<<GB(nc), 256, 0, stream>>>(aoutb, blin, dcat, nc);
  CA(dcat, bdcat, nc);
  gemm_rm(stream, bdcat, CW(154), P(155), dec1, 256, 512, 1024, 0);         // cat
  CA(dec1, bdec1, (long long)256 * 512);
  gemm_rm(stream, bdec1, CW(156), P(157), dec2, 256, 96, 512, 0);           // proj
  long long no = (long long)4 * 96 * 64;
  k_final<<<GB(no), 256, 0, stream>>>(dec2, stdv, means, (float*)d_out, no);
}
